// ChebConv_48679159332866
// MI455X (gfx1250) — matrix operand order repaired and checked by reading, not compiled, not run on hardware
//
#include <hip/hip_runtime.h>
#include <hip/hip_bf16.h>

// ---------------------------------------------------------------------------
// Problem constants (match reference)
// ---------------------------------------------------------------------------
constexpr int BATCH = 4;
constexpr int FIN   = 32;
constexpr int FOUT  = 64;
constexpr int NV    = 4096;
constexpr int NS    = 64;          // X*Y*Z = 4*4*4
constexpr int KCH   = 5;           // Chebyshev order
constexpr size_t ELEMS = (size_t)NV * BATCH * NS * FIN;   // 33,554,432 per cheb order
// cheb layout: cheb[k][v][b][s][fin], fin fastest (32), then s (64), then b (4), then v
// => per-v stride 8192, per-k stride ELEMS. Row for SpMM = 8192 contiguous elems.

typedef __attribute__((ext_vector_type(16))) __bf16 v16bf;
typedef __attribute__((ext_vector_type(8)))  float  v8f;

union V16 {
    v16bf          bf;
    uint4          u4[2];
    unsigned short us[16];
};

__device__ __forceinline__ unsigned short f2bf(float f) {
    unsigned int u = __float_as_uint(f);
    unsigned int r = u + 0x7FFFu + ((u >> 16) & 1u);   // round-to-nearest-even
    return (unsigned short)(r >> 16);
}
__device__ __forceinline__ float bf2f(unsigned int h) {
    return __uint_as_float(h << 16);
}

// ---------------------------------------------------------------------------
// 1) Reorder x [B,Fin,V,s] fp32 -> cheb0 [v][b][s][fin] bf16
// ---------------------------------------------------------------------------
__global__ __launch_bounds__(256) void cheb_reorder(const float* __restrict__ x,
                                                    unsigned short* __restrict__ cheb0) {
    size_t i = (size_t)blockIdx.x * 256 + threadIdx.x;       // over ELEMS
    int fin = (int)(i & 31);
    size_t j = i >> 5;
    int s = (int)(j & 63);  j >>= 6;
    int b = (int)(j & 3);
    size_t v = j >> 2;
    float xv = x[(((size_t)b * FIN + fin) * NV + v) * NS + s];
    cheb0[i] = f2bf(xv);
}

// ---------------------------------------------------------------------------
// 2a) Zero fp32 accumulator
// ---------------------------------------------------------------------------
__global__ __launch_bounds__(256) void zero_f32(float4* __restrict__ t) {
    size_t i = (size_t)blockIdx.x * 256 + threadIdx.x;       // over ELEMS/4
    t[i] = make_float4(0.f, 0.f, 0.f, 0.f);
}

// ---------------------------------------------------------------------------
// 2b) SpMM: t[rows[e], :] += vals[e] * cheb_prev[cols[e], :]
//     grid = (8, NNZ); each thread handles 4 consecutive features of one edge
// ---------------------------------------------------------------------------
__global__ __launch_bounds__(256) void cheb_spmm(const unsigned short* __restrict__ xprev,
                                                 float* __restrict__ t,
                                                 const int* __restrict__ rows,
                                                 const int* __restrict__ cols,
                                                 const float* __restrict__ vals) {
    int e  = blockIdx.y;
    int d0 = (blockIdx.x * 256 + threadIdx.x) * 4;           // 0..8188
    int r  = rows[e];
    int c  = cols[e];
    float val = vals[e];
    const unsigned short* xp = xprev + (size_t)c * 8192 + d0;
    uint2 raw = *(const uint2*)xp;                            // 4 bf16, 8B aligned
    float x0 = bf2f(raw.x & 0xFFFFu);
    float x1 = bf2f(raw.x >> 16);
    float x2 = bf2f(raw.y & 0xFFFFu);
    float x3 = bf2f(raw.y >> 16);
    float* tp = t + (size_t)r * 8192 + d0;
    unsafeAtomicAdd(tp + 0, val * x0);
    unsafeAtomicAdd(tp + 1, val * x1);
    unsafeAtomicAdd(tp + 2, val * x2);
    unsafeAtomicAdd(tp + 3, val * x3);
}

// ---------------------------------------------------------------------------
// 2c) Chebyshev combine: x_k = (doScale ? 2*t - x_{k-2} : t), store bf16
// ---------------------------------------------------------------------------
__global__ __launch_bounds__(256) void cheb_combine(const float* __restrict__ t,
                                                    const unsigned short* __restrict__ xm2,
                                                    unsigned short* __restrict__ xk,
                                                    int doScale) {
    size_t i = (size_t)blockIdx.x * 256 + threadIdx.x;       // over ELEMS
    float v = t[i];
    if (doScale) v = 2.0f * v - bf2f((unsigned int)xm2[i]);
    xk[i] = f2bf(v);
}

// ---------------------------------------------------------------------------
// 3) WMMA GEMM: out[n, fo] = sum_k sum_fin cheb[k][n][fin] * W[k][fin][fo] + bias
//    One wave per 16x64 output tile; 5 k-steps of 32; 4 N-tiles of 16.
//    Weights staged in LDS *transposed* [k][fo][fin] so each lane's B fragment
//    (16 halves at fixed fo, two 8-fin runs 16 apart) is two 16-byte LDS reads.
// ---------------------------------------------------------------------------
__global__ __launch_bounds__(256) void cheb_gemm_wmma(const unsigned short* __restrict__ cheb,
                                                      const float* __restrict__ weight,
                                                      const float* __restrict__ bias,
                                                      float* __restrict__ out) {
    __shared__ __align__(16) unsigned short wlds[KCH * FOUT * FIN]; // [k][fo][fin], 20 KB
    __shared__ __align__(32) float clds[8][FOUT][16];               // 8 waves * 4 KB = 32 KB

    int tid = threadIdx.x;
    // Convert + transpose weights fp32 [k][fin][fo] -> bf16 LDS [k][fo][fin]
    for (int i = tid; i < KCH * FIN * FOUT; i += 256) {
        int fo  = i & 63;
        int j   = i >> 6;
        int fin = j & 31;
        int k   = j >> 5;
        wlds[(k * FOUT + fo) * FIN + fin] = f2bf(weight[i]);
    }
    __syncthreads();

    int wave = tid >> 5;
    int lane = tid & 31;
    int laneLo = lane & 15;
    int laneHi = lane >> 4;                                   // 0 or 1

    // tile id -> (batch, vertex, s0)
    int tile = blockIdx.x * 8 + wave;                         // 65536 tiles total
    int bb   = tile >> 14;                                    // / (V*4)
    int rem  = tile & 16383;
    int v    = rem >> 2;
    int s0   = (rem & 3) * 16;

    v8f acc[4];
    #pragma unroll
    for (int nt = 0; nt < 4; ++nt)
        #pragma unroll
        for (int r = 0; r < 8; ++r) acc[nt][r] = 0.0f;

    // A fragment base: lane row s = s0 + laneLo; fin half-group = laneHi*8
    size_t abase = (size_t)v * 8192 + (size_t)bb * 2048
                 + (size_t)(s0 + laneLo) * 32 + (size_t)(laneHi * 8);
    // B fragment: lane group selects fin group (laneHi*8), fo = nt*16 + laneLo
    int finb = laneHi * 8;

    #pragma unroll
    for (int k = 0; k < KCH; ++k) {
        V16 a;
        const uint4* ap = (const uint4*)(cheb + (size_t)k * ELEMS + abase);
        a.u4[0] = ap[0];          // fins finb8 .. finb8+7
        a.u4[1] = ap[2];          // fins finb8+16 .. finb8+23   (+16 elems = 2 uint4)

        #pragma unroll
        for (int nt = 0; nt < 4; ++nt) {
            V16 bw;
            const uint4* bp =
                (const uint4*)&wlds[(k * FOUT + nt * 16 + laneLo) * FIN + finb];
            bw.u4[0] = bp[0];     // fins finb .. finb+7  (16 B contiguous)
            bw.u4[1] = bp[2];     // fins finb+16 .. finb+23
            acc[nt] = __builtin_amdgcn_wmma_f32_16x16x32_bf16(
                false, a.bf, false, bw.bf, (short)0, acc[nt], false, false);
        }
    }

    // Repack C through LDS: clds[wave][fo_local][s_local]; one 32B store per N-tile
    #pragma unroll
    for (int nt = 0; nt < 4; ++nt)
        *(v8f*)&clds[wave][nt * 16 + laneLo][laneHi * 8] = acc[nt];
    __syncthreads();

    // Coalesced stores: out[((b*FOUT+fo)*V + v)*NS + s0 + s_local], float4 chunks
    size_t outbase = (((size_t)bb * FOUT) * NV + v) * NS + s0;
    #pragma unroll
    for (int it = 0; it < 8; ++it) {
        int chunk = it * 32 + lane;                           // 0..255
        int fo    = chunk >> 2;
        int part  = chunk & 3;
        float4 val = *(const float4*)&clds[wave][fo][part * 4];
        float  bv  = bias[fo];
        val.x += bv; val.y += bv; val.z += bv; val.w += bv;
        *(float4*)(out + outbase + (size_t)fo * NV * NS + part * 4) = val;
    }
}

// ---------------------------------------------------------------------------
// Launcher
// ---------------------------------------------------------------------------
extern "C" void kernel_launch(void* const* d_in, const int* in_sizes, int n_in,
                              void* d_out, int out_size, void* d_ws, size_t ws_size,
                              hipStream_t stream) {
    const float* x      = (const float*)d_in[0];
    const float* vals   = (const float*)d_in[1];
    const float* weight = (const float*)d_in[2];
    const float* bias   = (const float*)d_in[3];
    const int*   rows   = (const int*)d_in[4];
    const int*   cols   = (const int*)d_in[5];
    const int nnz = in_sizes[1];                              // 28672

    // Workspace: [5 * ELEMS bf16 cheb][ELEMS fp32 accumulator] = ~470 MB
    unsigned short* cheb = (unsigned short*)d_ws;
    float* t = (float*)((char*)d_ws + (size_t)KCH * ELEMS * sizeof(unsigned short));

    const int nblk_elem = (int)(ELEMS / 256);                 // 131072

    // x0 = reordered input
    cheb_reorder<<<nblk_elem, 256, 0, stream>>>(x, cheb);

    // Chebyshev recurrence: x1 = L x0; x_k = 2 L x_{k-1} - x_{k-2}
    for (int k = 1; k < KCH; ++k) {
        zero_f32<<<(int)(ELEMS / 4 / 256), 256, 0, stream>>>((float4*)t);
        cheb_spmm<<<dim3(8, nnz), 256, 0, stream>>>(
            cheb + (size_t)(k - 1) * ELEMS, t, rows, cols, vals);
        cheb_combine<<<nblk_elem, 256, 0, stream>>>(
            t,
            cheb + (size_t)(k >= 2 ? (k - 2) : 0) * ELEMS,
            cheb + (size_t)k * ELEMS,
            k >= 2 ? 1 : 0);
    }

    // Fused GEMM + bias + output layout
    const int ntiles = BATCH * NV * (NS / 16);                // 65536
    cheb_gemm_wmma<<<ntiles / 8, 256, 0, stream>>>(cheb, weight, bias, (float*)d_out);
}
